// HeteroGraphConv_52974126629629
// MI455X (gfx1250) — compile-verified, matrix operand order checked by reading, not run on hardware
//
#include <hip/hip_runtime.h>
#include <hip/hip_bf16.h>

#define H       256
#define NOP     200000
#define NTAB    50000
#define NCOL    100000
#define ECALL   400000
#define ESCAN   100000
#define EOUT    200000
#define NGRAPH  4096
#define NLAYERS 3

typedef __bf16 bf16_t;
typedef __attribute__((ext_vector_type(16))) __bf16 v16bf;
typedef __attribute__((ext_vector_type(8)))  __bf16 v8bf;
typedef __attribute__((ext_vector_type(4)))  __bf16 v4bf;
typedef __attribute__((ext_vector_type(8)))  float  v8f;

// ---------------------------------------------------------------------------
// Encoder: Y[M x 256] = X[M x K] @ W[K x 256] + b   (no activation)
// ---------------------------------------------------------------------------
template<int K>
__global__ __launch_bounds__(256) void encode_kernel(
    const float* __restrict__ X, const float* __restrict__ W,
    const float* __restrict__ b, float* __restrict__ Y, int M) {
  __shared__ float xs[K];
  const int row = blockIdx.x;
  const int t = threadIdx.x;
  if (t < K) xs[t] = X[(size_t)row * K + t];
  __syncthreads();
  float acc = b[t];
#pragma unroll
  for (int k = 0; k < K; ++k) acc += xs[k] * W[k * H + t];
  Y[(size_t)row * H + t] = acc;
}

// ---------------------------------------------------------------------------
// Pre-sum + convert weights to bf16, stored TRANSPOSED: Wt[n*256 + k].
// ---------------------------------------------------------------------------
__global__ void prep_weights_kernel(
    const float* __restrict__ wr_c, const float* __restrict__ wr_s, const float* __restrict__ wr_o,
    const float* __restrict__ wt_c, const float* __restrict__ wt_s, const float* __restrict__ wt_o,
    const float* __restrict__ wr_tl, const float* __restrict__ wt_tl,
    const float* __restrict__ wr_cl, const float* __restrict__ wt_cl,
    const float* __restrict__ b_c, const float* __restrict__ b_s, const float* __restrict__ b_o,
    bf16_t* __restrict__ w1, bf16_t* __restrict__ w2, bf16_t* __restrict__ w3,
    bf16_t* __restrict__ wroot, bf16_t* __restrict__ wt, bf16_t* __restrict__ wc,
    float* __restrict__ b_sum) {
  const int i = blockIdx.x * blockDim.x + threadIdx.x;   // i = k*256 + n
  if (i < H * H) {
    const int k = i >> 8;
    const int n = i & 255;
    const int o = n * H + k;                              // transposed index
    w1[o]    = (bf16_t)wr_c[i];
    w2[o]    = (bf16_t)wr_s[i];
    w3[o]    = (bf16_t)wr_o[i];
    wroot[o] = (bf16_t)(wt_c[i] + wt_s[i] + wt_o[i]);
    wt[o]    = (bf16_t)(wr_tl[i] + wt_tl[i]);
    wc[o]    = (bf16_t)(wr_cl[i] + wt_cl[i]);
  }
  if (i < H) b_sum[i] = b_c[i] + b_s[i] + b_o[i];
}

// ---------------------------------------------------------------------------
// Edge scatter-add: agg[dst] += X[src] per edge (float4 per thread).
// ---------------------------------------------------------------------------
__global__ void scatter_add_kernel(float* __restrict__ agg, const float* __restrict__ X,
                                   const int* __restrict__ src, const int* __restrict__ dst,
                                   int nE) {
  const long tid = (long)blockIdx.x * blockDim.x + threadIdx.x;
  const long total = (long)nE * (H / 4);
  if (tid >= total) return;
  const int e = (int)(tid >> 6);
  const int c = (int)(tid & 63);
  const int s = src[e], d = dst[e];
  const float4 v = ((const float4*)(X + (size_t)s * H))[c];
  float* o = agg + (size_t)d * H + (size_t)c * 4;
  unsafeAtomicAdd(o + 0, v.x);
  unsafeAtomicAdd(o + 1, v.y);
  unsafeAtomicAdd(o + 2, v.z);
  unsafeAtomicAdd(o + 3, v.w);
}

// ---------------------------------------------------------------------------
// Fused multi-source GEMM + bias + ReLU, bf16 WMMA with f32 accumulation.
//   Y[M x 256] = relu( sum_{r<NR} A_r[M x 256] @ W_r + bias )
// W_r supplied transposed bf16 (Wt[n][k]). Block = 256 thr (8 wave32),
// tile 128x128, wave = 32x64 = 2x4 wmma accs. Fully unrolled step loop,
// DOUBLE-BUFFERED LDS + register prefetch: one barrier per step, the
// global-load wait sits after the step's 8 WMMAs.
// ---------------------------------------------------------------------------
#define LDT 40  // LDS row stride in bf16 (80 B) -> conflict-free b128 frags

template<int NR>
__global__ __launch_bounds__(256) void gemm_relu_kernel(
    const float* __restrict__ A0, const float* __restrict__ A1,
    const float* __restrict__ A2, const float* __restrict__ A3,
    const bf16_t* __restrict__ W0, const bf16_t* __restrict__ W1,
    const bf16_t* __restrict__ W2, const bf16_t* __restrict__ W3,
    const float* __restrict__ bias, float* __restrict__ Y, int M) {
  __shared__ __align__(16) bf16_t As[2][128 * LDT];
  __shared__ __align__(16) bf16_t Bs[2][128 * LDT];

  const int rowBase = blockIdx.x * 128;
  const int colBase = blockIdx.y * 128;
  const int t    = threadIdx.x;
  const int lane = t & 31;
  const int wid  = t >> 5;
  const int wm   = (wid & 3) * 32;   // wave row offset in tile
  const int wn   = (wid >> 2) * 64;  // wave col offset in tile
  const int lr   = lane & 15;
  const int kh   = lane >> 4;        // K-half per ISA 16-bit A/B layout

  // per-thread staging geometry (constant)
  const int a_row0 = t >> 3;           // + 32*j
  const int a_c4   = (t & 7) * 4;
  const int b_n0   = t >> 2;           // + 64*j
  const int b_kb   = (t & 3) * 8;

  v8f acc[2][4];
#pragma unroll
  for (int i = 0; i < 2; ++i)
#pragma unroll
    for (int j = 0; j < 4; ++j)
#pragma unroll
      for (int e = 0; e < 8; ++e) acc[i][j][e] = 0.0f;

  float4 aReg[4];
  v8bf   bReg[2];

  auto loadA = [&](const float* __restrict__ A, int k0) {
#pragma unroll
    for (int j = 0; j < 4; ++j) {
      int grow = rowBase + a_row0 + 32 * j;     // branch-free: clamp row
      grow = grow < M ? grow : M - 1;
      aReg[j] = *(const float4*)(A + (size_t)grow * H + k0 + a_c4);
    }
  };
  auto loadB = [&](const bf16_t* __restrict__ Wt, int k0) {
#pragma unroll
    for (int j = 0; j < 2; ++j) {
      const int n = b_n0 + 64 * j;
      bReg[j] = *(const v8bf*)(Wt + (size_t)(colBase + n) * H + k0 + b_kb);
    }
  };
  auto storeLDS = [&](int buf) {
#pragma unroll
    for (int j = 0; j < 4; ++j) {
      v4bf pk;
      pk[0] = (bf16_t)aReg[j].x; pk[1] = (bf16_t)aReg[j].y;
      pk[2] = (bf16_t)aReg[j].z; pk[3] = (bf16_t)aReg[j].w;
      *(v4bf*)&As[buf][(a_row0 + 32 * j) * LDT + a_c4] = pk;
    }
#pragma unroll
    for (int j = 0; j < 2; ++j)
      *(v8bf*)&Bs[buf][(b_n0 + 64 * j) * LDT + b_kb] = bReg[j];
  };
  auto compute = [&](int buf) {
    v16bf af[2];
#pragma unroll
    for (int rt = 0; rt < 2; ++rt) {
      const bf16_t* base = &As[buf][(wm + rt * 16 + lr) * LDT + kh * 8];
      v8bf lo = *(const v8bf*)base;
      v8bf hi = *(const v8bf*)(base + 16);
      af[rt] = __builtin_shufflevector(lo, hi, 0, 1, 2, 3, 4, 5, 6, 7,
                                       8, 9, 10, 11, 12, 13, 14, 15);
    }
    v16bf bfr[4];
#pragma unroll
    for (int ct = 0; ct < 4; ++ct) {
      const bf16_t* base = &Bs[buf][(wn + ct * 16 + lr) * LDT + kh * 8];
      v8bf lo = *(const v8bf*)base;
      v8bf hi = *(const v8bf*)(base + 16);
      bfr[ct] = __builtin_shufflevector(lo, hi, 0, 1, 2, 3, 4, 5, 6, 7,
                                        8, 9, 10, 11, 12, 13, 14, 15);
    }
#pragma unroll
    for (int rt = 0; rt < 2; ++rt)
#pragma unroll
      for (int ct = 0; ct < 4; ++ct)
        acc[rt][ct] = __builtin_amdgcn_wmma_f32_16x16x32_bf16(
            false, af[rt], false, bfr[ct], (short)0, acc[rt][ct],
            false, false);
  };

  const float*  Aarr[4] = {A0, A1, A2, A3};
  const bf16_t* Warr[4] = {W0, W1, W2, W3};

  constexpr int NSTEP = NR * 8;
  // prologue: fetch + stage step 0 into buffer 0
  loadA(Aarr[0], 0);
  loadB(Warr[0], 0);
  storeLDS(0);
  __syncthreads();

#pragma unroll
  for (int s = 0; s < NSTEP; ++s) {
    if (s + 1 < NSTEP) {                 // issue next step's global loads
      const int r  = (s + 1) >> 3;       // constant after full unroll
      const int k0 = ((s + 1) & 7) * 32;
      loadA(Aarr[r], k0);
      loadB(Warr[r], k0);
    }
    compute(s & 1);                      // 8 WMMAs hide the load latency
    if (s + 1 < NSTEP) {
      storeLDS((s + 1) & 1);             // other buffer: no read conflict
      __syncthreads();                   // single barrier per step
    }
  }

  // ---- epilogue: bias + ReLU. C/D layout: lane%16 = col, vgpr r + 8*kh = row
#pragma unroll
  for (int rt = 0; rt < 2; ++rt) {
#pragma unroll
    for (int ct = 0; ct < 4; ++ct) {
      const int col = colBase + wn + ct * 16 + lr;
      const float bv = bias[col];
#pragma unroll
      for (int rr = 0; rr < 8; ++rr) {
        const int row = rowBase + wm + rt * 16 + rr + 8 * kh;
        if (row < M) {
          const float v = acc[rt][ct][rr] + bv;
          Y[(size_t)row * H + col] = v > 0.f ? v : 0.f;
        }
      }
    }
  }
}

// ---------------------------------------------------------------------------
// Mean-pool accumulation over operator nodes per graph.
// ---------------------------------------------------------------------------
__global__ void pool_accum_kernel(const float* __restrict__ xo, const int* __restrict__ batch,
                                  float* __restrict__ pooled, float* __restrict__ cnt) {
  const long tid = (long)blockIdx.x * blockDim.x + threadIdx.x;
  const long total = (long)NOP * (H / 4);
  if (tid >= total) return;
  const int node = (int)(tid >> 6);
  const int c = (int)(tid & 63);
  const int g = batch[node];
  const float4 v = ((const float4*)(xo + (size_t)node * H))[c];
  float* o = pooled + (size_t)g * H + (size_t)c * 4;
  unsafeAtomicAdd(o + 0, v.x);
  unsafeAtomicAdd(o + 1, v.y);
  unsafeAtomicAdd(o + 2, v.z);
  unsafeAtomicAdd(o + 3, v.w);
  if (c == 0) unsafeAtomicAdd(cnt + g, 1.0f);
}

// ---------------------------------------------------------------------------
// Heads: out[g] = pooled/cnt . w_mem + b_mem ; out[NG+g] = ... w_time + b_time
// ---------------------------------------------------------------------------
__global__ __launch_bounds__(256) void head_kernel(
    const float* __restrict__ pooled, const float* __restrict__ cnt,
    const float* __restrict__ w_mem, const float* __restrict__ b_mem,
    const float* __restrict__ w_time, const float* __restrict__ b_time,
    float* __restrict__ out) {
  __shared__ float sm[256];
  __shared__ float st[256];
  const int g = blockIdx.x, t = threadIdx.x;
  float c = cnt[g];
  c = c > 1.0f ? c : 1.0f;
  const float v = pooled[(size_t)g * H + t] / c;
  sm[t] = v * w_mem[t];
  st[t] = v * w_time[t];
  __syncthreads();
  for (int s = 128; s > 0; s >>= 1) {
    if (t < s) { sm[t] += sm[t + s]; st[t] += st[t + s]; }
    __syncthreads();
  }
  if (t == 0) {
    out[g] = sm[0] + b_mem[0];
    out[NGRAPH + g] = st[0] + b_time[0];
  }
}

// ---------------------------------------------------------------------------
extern "C" void kernel_launch(void* const* d_in, const int* in_sizes, int n_in,
                              void* d_out, int out_size, void* d_ws, size_t ws_size,
                              hipStream_t stream) {
  const float* x_op        = (const float*)d_in[0];
  const float* x_tab       = (const float*)d_in[1];
  const float* x_col       = (const float*)d_in[2];
  const int*   e_called    = (const int*)d_in[3];   // [2, ECALL]: row0=src, row1=dst
  const int*   e_scan_src  = (const int*)d_in[4];
  const int*   e_scan_dst  = (const int*)d_in[5];
  const int*   e_out_src   = (const int*)d_in[6];
  const int*   e_out_dst   = (const int*)d_in[7];
  const int*   batch       = (const int*)d_in[8];
  const float* w_op        = (const float*)d_in[9];
  const float* b_op        = (const float*)d_in[10];
  const float* w_tab       = (const float*)d_in[11];
  const float* b_tab       = (const float*)d_in[12];
  const float* w_col       = (const float*)d_in[13];
  const float* b_col       = (const float*)d_in[14];
  const float* w_rel_c     = (const float*)d_in[15];
  const float* b_rel_c     = (const float*)d_in[16];
  const float* w_root_c    = (const float*)d_in[17];
  const float* w_rel_s     = (const float*)d_in[18];
  const float* b_rel_s     = (const float*)d_in[19];
  const float* w_root_s    = (const float*)d_in[20];
  const float* w_rel_o     = (const float*)d_in[21];
  const float* b_rel_o     = (const float*)d_in[22];
  const float* w_root_o    = (const float*)d_in[23];
  const float* w_rel_tl    = (const float*)d_in[24];
  const float* b_rel_tl    = (const float*)d_in[25];
  const float* w_root_tl   = (const float*)d_in[26];
  const float* w_rel_cl    = (const float*)d_in[27];
  const float* b_rel_cl    = (const float*)d_in[28];
  const float* w_root_cl   = (const float*)d_in[29];
  const float* w_mem       = (const float*)d_in[30];
  const float* b_mem       = (const float*)d_in[31];
  const float* w_time      = (const float*)d_in[32];
  const float* b_time      = (const float*)d_in[33];

  // ---- workspace carve (needed ~1.35 GB)
  char* p = (char*)d_ws;
  auto carve = [&](size_t bytes) {
    char* q = p;
    p += (bytes + 255) & ~(size_t)255;
    return q;
  };
  const size_t NOPH  = (size_t)NOP * H * sizeof(float);
  const size_t NTABH = (size_t)NTAB * H * sizeof(float);
  const size_t NCOLH = (size_t)NCOL * H * sizeof(float);
  float* xo_a   = (float*)carve(NOPH);
  float* xo_b   = (float*)carve(NOPH);
  float* xt_a   = (float*)carve(NTABH);
  float* xt_b   = (float*)carve(NTABH);
  float* xc_a   = (float*)carve(NCOLH);
  float* xc_b   = (float*)carve(NCOLH);
  float* agg_c  = (float*)carve(NOPH);
  float* agg_s  = (float*)carve(NOPH);
  float* agg_o  = (float*)carve(NOPH);
  float* pooled = (float*)carve((size_t)NGRAPH * H * sizeof(float));
  float* cntbuf = (float*)carve((size_t)NGRAPH * sizeof(float));
  bf16_t* w1    = (bf16_t*)carve((size_t)H * H * sizeof(bf16_t));
  bf16_t* w2    = (bf16_t*)carve((size_t)H * H * sizeof(bf16_t));
  bf16_t* w3    = (bf16_t*)carve((size_t)H * H * sizeof(bf16_t));
  bf16_t* wroot = (bf16_t*)carve((size_t)H * H * sizeof(bf16_t));
  bf16_t* wt    = (bf16_t*)carve((size_t)H * H * sizeof(bf16_t));
  bf16_t* wc    = (bf16_t*)carve((size_t)H * H * sizeof(bf16_t));
  float* b_sum  = (float*)carve((size_t)H * sizeof(float));

  // ---- one-time weight prep (bf16 conversion + pre-sums + transpose)
  prep_weights_kernel<<<(H * H + 255) / 256, 256, 0, stream>>>(
      w_rel_c, w_rel_s, w_rel_o, w_root_c, w_root_s, w_root_o,
      w_rel_tl, w_root_tl, w_rel_cl, w_root_cl,
      b_rel_c, b_rel_s, b_rel_o,
      w1, w2, w3, wroot, wt, wc, b_sum);

  // ---- input encoders
  encode_kernel<32><<<NOP, 256, 0, stream>>>(x_op, w_op, b_op, xo_a, NOP);
  encode_kernel<16><<<NTAB, 256, 0, stream>>>(x_tab, w_tab, b_tab, xt_a, NTAB);
  encode_kernel<16><<<NCOL, 256, 0, stream>>>(x_col, w_col, b_col, xc_a, NCOL);

  float* xo_in = xo_a; float* xo_out = xo_b;
  float* xt_in = xt_a; float* xt_out = xt_b;
  float* xc_in = xc_a; float* xc_out = xc_b;

  const dim3 gemm_block(256);
  for (int l = 0; l < NLAYERS; ++l) {
    hipMemsetAsync(agg_c, 0, NOPH, stream);
    hipMemsetAsync(agg_s, 0, NOPH, stream);
    hipMemsetAsync(agg_o, 0, NOPH, stream);

    scatter_add_kernel<<<(int)(((long)ECALL * 64 + 255) / 256), 256, 0, stream>>>(
        agg_c, xo_in, e_called, e_called + ECALL, ECALL);
    scatter_add_kernel<<<(int)(((long)ESCAN * 64 + 255) / 256), 256, 0, stream>>>(
        agg_s, xt_in, e_scan_src, e_scan_dst, ESCAN);
    scatter_add_kernel<<<(int)(((long)EOUT * 64 + 255) / 256), 256, 0, stream>>>(
        agg_o, xc_in, e_out_src, e_out_dst, EOUT);

    gemm_relu_kernel<4><<<dim3((NOP + 127) / 128, 2), gemm_block, 0, stream>>>(
        agg_c, agg_s, agg_o, xo_in, w1, w2, w3, wroot, b_sum, xo_out, NOP);
    gemm_relu_kernel<1><<<dim3((NTAB + 127) / 128, 2), gemm_block, 0, stream>>>(
        xt_in, nullptr, nullptr, nullptr, wt, nullptr, nullptr, nullptr,
        b_rel_tl, xt_out, NTAB);
    gemm_relu_kernel<1><<<dim3((NCOL + 127) / 128, 2), gemm_block, 0, stream>>>(
        xc_in, nullptr, nullptr, nullptr, wc, nullptr, nullptr, nullptr,
        b_rel_cl, xc_out, NCOL);

    float* tmp;
    tmp = xo_in; xo_in = xo_out; xo_out = tmp;
    tmp = xt_in; xt_in = xt_out; xt_out = tmp;
    tmp = xc_in; xc_in = xc_out; xc_out = tmp;
  }

  // ---- global mean pool + heads
  hipMemsetAsync(pooled, 0, (size_t)NGRAPH * H * sizeof(float), stream);
  hipMemsetAsync(cntbuf, 0, (size_t)NGRAPH * sizeof(float), stream);
  pool_accum_kernel<<<(int)(((long)NOP * 64 + 255) / 256), 256, 0, stream>>>(
      xo_in, batch, pooled, cntbuf);
  head_kernel<<<NGRAPH, 256, 0, stream>>>(pooled, cntbuf, w_mem, b_mem,
                                          w_time, b_time, (float*)d_out);
}
